// BagOfWords_30262339568354
// MI455X (gfx1250) — compile-verified, matrix operand order checked by reading, not run on hardware
//
#include <hip/hip_runtime.h>

// ---------------------------------------------------------------------------
// B=8, T=1024, V=8000, D=384
// out[b,t,v] = (cumsum_t((t+1)*emb[ctx]) / den[t]) . W[v,:] + bias[v]
// Restructured: prefix-scan in D-space (tiny), then a single bf16-WMMA GEMM
// M=8192 x N=8000 x K=384 with f32 accumulation.
// ---------------------------------------------------------------------------

#define BB 8
#define TT 1024
#define VV 8000
#define DD 384

typedef __attribute__((ext_vector_type(16))) __bf16 v16bf;
typedef __attribute__((ext_vector_type(8)))  float  v8f;

__device__ __forceinline__ unsigned short f2bf(float f) {
  unsigned u = __float_as_uint(f);
  u += 0x7FFFu + ((u >> 16) & 1u);   // round-to-nearest-even
  return (unsigned short)(u >> 16);
}

// ---- Kernel 1: convert W (V*D f32) -> bf16 ---------------------------------
__global__ void convert_w_kernel(const float* __restrict__ W,
                                 unsigned short* __restrict__ Wb, int n) {
  int i = blockIdx.x * blockDim.x + threadIdx.x;
  if (i < n) Wb[i] = f2bf(W[i]);
}

// ---- Kernel 2: gather + weighted causal prefix scan, normalized, -> bf16 ---
// Q[b,t,d] = (sum_{s<=t} (s+1)*emb[ctx[b,s],d]) / ((t+1)(t+2)/2)
// One block = (batch b, 128-channel chunk of D). Token ids staged in LDS so
// all global loads in the scan have addresses known up front (pipelineable).
__global__ __launch_bounds__(128) void scan_kernel(
    const int* __restrict__ ctx, const float* __restrict__ emb,
    unsigned short* __restrict__ Qb) {
  __shared__ int s_tok[TT];
  const int b  = blockIdx.x / (DD / 128);
  const int d  = (blockIdx.x % (DD / 128)) * 128 + threadIdx.x;
  for (int i = threadIdx.x; i < TT; i += 128) s_tok[i] = ctx[b * TT + i];
  __syncthreads();
  float acc = 0.0f;
#pragma unroll 8
  for (int t = 0; t < TT; ++t) {
    const int tok = s_tok[t];
    const float v = emb[(size_t)tok * DD + d];
    acc = fmaf((float)(t + 1), v, acc);
    const float inv_den = 2.0f / ((float)(t + 1) * (float)(t + 2));
    Qb[((size_t)b * TT + t) * DD + d] = f2bf(acc * inv_den);
  }
}

// ---- Kernel 3: GEMM out[M,N] = Q[M,K] * Wb[N,K]^T + bias, f32 out ----------
// WG = 256 threads = 8 waves, WG tile 128x128, wave tile 64(M) x 32(N).
// Fragment layout (16-bit A, 16x32): lane<16 -> row=lane, K chunks {0..7,16..23};
// lane>=16 -> row=lane-16, K chunks {8..15,24..31}. B uses the mirrored layout
// with N in place of M, so loading W rows (Bt row-major) is symmetric.
__global__ __launch_bounds__(256) void gemm_kernel(
    const unsigned short* __restrict__ Qb,
    const unsigned short* __restrict__ Wb,
    const float* __restrict__ bias,
    float* __restrict__ out) {
  const int lane  = threadIdx.x & 31;
  const int wave  = threadIdx.x >> 5;
  const int wm    = wave & 1;        // 2 waves along M (2*64 = 128)
  const int wn    = wave >> 1;       // 4 waves along N (4*32 = 128)
  const int l15   = lane & 15;
  const int lhalf = lane >> 4;       // 0/1: which K-chunk pair this lane holds

  const int wgRow = blockIdx.y * 128;
  const int wgCol = blockIdx.x * 128;

  // A fragment base pointers (one 16-row tile each), advanced by 32 elems/step
  const unsigned short* aP[4];
#pragma unroll
  for (int mi = 0; mi < 4; ++mi) {
    const int row = wgRow + wm * 64 + mi * 16 + l15;      // M=8192, always valid
    aP[mi] = Qb + (size_t)row * DD + lhalf * 8;
  }
  // B fragment base pointers; clamp OOB columns (results discarded on store)
  const unsigned short* bP[2];
  int cols[2];
#pragma unroll
  for (int ni = 0; ni < 2; ++ni) {
    const int col = wgCol + wn * 32 + ni * 16 + l15;
    cols[ni] = col;
    const int cc = col < VV ? col : (VV - 1);
    bP[ni] = Wb + (size_t)cc * DD + lhalf * 8;
  }

  v8f acc[4][2] = {};

  union Frag { int4 q[2]; v16bf v; };

  for (int k = 0; k < DD; k += 32) {
    Frag a[4], bf[2];
#pragma unroll
    for (int mi = 0; mi < 4; ++mi) {
      a[mi].q[0] = *(const int4*)(aP[mi]);        // K = klo .. klo+7
      a[mi].q[1] = *(const int4*)(aP[mi] + 16);   // K = klo+16 .. klo+23
      aP[mi] += 32;
    }
#pragma unroll
    for (int ni = 0; ni < 2; ++ni) {
      bf[ni].q[0] = *(const int4*)(bP[ni]);
      bf[ni].q[1] = *(const int4*)(bP[ni] + 16);
      bP[ni] += 32;
    }
#pragma unroll
    for (int mi = 0; mi < 4; ++mi)
#pragma unroll
      for (int ni = 0; ni < 2; ++ni)
        acc[mi][ni] = __builtin_amdgcn_wmma_f32_16x16x32_bf16(
            false, a[mi].v, false, bf[ni].v, (short)0, acc[mi][ni],
            false, false);
  }

  // Epilogue: C/D layout — VGPR r: lanes 0-15 -> M=r, lanes 16-31 -> M=r+8;
  // N = lane & 15. Add bias, guard N bounds, write f32.
#pragma unroll
  for (int ni = 0; ni < 2; ++ni) {
    const int col = cols[ni];
    if (col >= VV) continue;
    const float bv = bias[col];
#pragma unroll
    for (int mi = 0; mi < 4; ++mi) {
      const int rowBase = wgRow + wm * 64 + mi * 16 + 8 * lhalf;
#pragma unroll
      for (int r = 0; r < 8; ++r) {
        out[(size_t)(rowBase + r) * VV + col] = acc[mi][ni][r] + bv;
      }
    }
  }
}

// ---------------------------------------------------------------------------
extern "C" void kernel_launch(void* const* d_in, const int* in_sizes, int n_in,
                              void* d_out, int out_size, void* d_ws, size_t ws_size,
                              hipStream_t stream) {
  const int*   ctx  = (const int*)d_in[0];    // (B,T) int32
  const float* emb  = (const float*)d_in[1];  // (V,D) f32
  const float* W    = (const float*)d_in[2];  // (V,D) f32
  const float* bias = (const float*)d_in[3];  // (V,)  f32
  float* out = (float*)d_out;                 // (B,T,V) f32

  unsigned short* Qb = (unsigned short*)d_ws;              // 8192*384 bf16
  unsigned short* Wb = Qb + (size_t)BB * TT * DD;          // 8000*384 bf16

  // 1) W -> bf16
  {
    const int n = VV * DD;
    convert_w_kernel<<<(n + 255) / 256, 256, 0, stream>>>(W, Wb, n);
  }
  // 2) gather + weighted prefix scan -> normalized bf16 Q
  scan_kernel<<<BB * (DD / 128), 128, 0, stream>>>(ctx, emb, Qb);
  // 3) bf16 WMMA GEMM with bias epilogue
  {
    dim3 grid((VV + 127) / 128, (BB * TT) / 128);  // 63 x 64
    gemm_kernel<<<grid, 256, 0, stream>>>(Qb, Wb, bias, out);
  }
}
